// SVDImageFunction_60928406061701
// MI455X (gfx1250) — matrix operand order repaired and checked by reading, not compiled, not run on hardware
//
#include <hip/hip_runtime.h>

// Problem constants (from the reference)
#define RES   2048
#define RANK  512
#define KS    51
#define HALO  25          // KS/2
#define OFS   (3u * RES * RANK)   // elements per f16 workspace array (3,145,728)

typedef _Float16 v16h __attribute__((ext_vector_type(16)));
typedef _Float16 v8h  __attribute__((ext_vector_type(8)));
typedef float    v8f  __attribute__((ext_vector_type(8)));

// CDNA5 async memory->LDS copy (16B per lane), tracked by ASYNCcnt.
// vdst = VGPR with LDS byte address, vaddr = 64-bit global address.
#define ASYNC_CP16(ldsoff, gptr)                                            \
    asm volatile("global_load_async_to_lds_b128 %0, %1, off"                \
                 :: "v"(ldsoff), "v"(gptr) : "memory")

#if __has_builtin(__builtin_amdgcn_s_wait_asynccnt)
#define WAIT_ASYNC(n) __builtin_amdgcn_s_wait_asynccnt(n)
#else
#define WAIT_ASYNC(n) asm volatile("s_wait_asynccnt %0" :: "i"(n) : "memory")
#endif

// ---------------------------------------------------------------------------
// Kernel 1: fused 51-tap blur + fp32->f16 hi/lo split + transpose.
// Input:  x[c][r][pos] fp32   (r1 or r2 per blockIdx.z)
// Output: hi/lo f16 arrays laid out [c][pos][r]  (K=r innermost -> WMMA-friendly)
// Block = 256 threads processes a 64(r) x 64(pos) tile.
// ---------------------------------------------------------------------------
__global__ __launch_bounds__(256) void blur_split_transpose_kernel(
    const float* __restrict__ rank1, const float* __restrict__ rank2,
    const float* __restrict__ kern, _Float16* __restrict__ ws)
{
    __shared__ float s_tile[64 * 115];   // 64 rows x (64+50) cols, stride 115 (odd -> bank friendly)
    __shared__ float s_k[KS];

    const int tid      = threadIdx.x;
    const int posChunk = blockIdx.x;        // 0..31
    const int c        = blockIdx.y >> 3;   // 0..2
    const int rChunk   = blockIdx.y & 7;    // 0..7
    const int src      = blockIdx.z;        // 0: rank2 -> A arrays, 1: rank1 -> B arrays
    const int pos0     = posChunk * 64;
    const int r0       = rChunk * 64;

    const float* in = src ? rank1 : rank2;
    _Float16* outHi = ws + (size_t)src * 2u * OFS;   // A:{0,OFS}  B:{2*OFS,3*OFS}
    _Float16* outLo = outHi + OFS;

    if (tid < KS) s_k[tid] = kern[tid];

    // Load 64 x 114 input window (zero-padded at the pos edges)
    for (int idx = tid; idx < 64 * 114; idx += 256) {
        int rl  = idx / 114;
        int off = idx - rl * 114;
        int gp  = pos0 - HALO + off;
        float v = 0.0f;
        if (gp >= 0 && gp < RES)
            v = in[((size_t)(c * RANK + r0 + rl)) * RES + gp];
        s_tile[rl * 115 + off] = v;
    }
    __syncthreads();

    // Each thread: one pos column, 16 r rows -> 16 contiguous halfs in the
    // transposed output (vectorized 16B stores).
    const int pl = tid >> 2;   // 0..63  local pos
    const int rg = tid & 3;    // 0..3   r group (16 rows each)

    v8h hi0, hi1, lo0, lo1;
    #pragma unroll
    for (int i = 0; i < 16; ++i) {
        int rl = rg * 16 + i;
        const float* row = &s_tile[rl * 115 + pl];
        float s = 0.0f;
        #pragma unroll
        for (int j = 0; j < KS; ++j) s = fmaf(s_k[j], row[j], s);
        _Float16 h = (_Float16)s;
        _Float16 l = (_Float16)(s - (float)h);   // exact residual -> ~fp32 via 3 WMMAs
        if (i < 8) { hi0[i] = h; lo0[i] = l; }
        else       { hi1[i - 8] = h; lo1[i - 8] = l; }
    }

    size_t ob = ((size_t)(c * RES + pos0 + pl)) * RANK + r0 + rg * 16;
    *(v8h*)&outHi[ob]     = hi0;
    *(v8h*)&outHi[ob + 8] = hi1;
    *(v8h*)&outLo[ob]     = lo0;
    *(v8h*)&outLo[ob + 8] = lo1;
}

// ---------------------------------------------------------------------------
// Kernel 2: per-channel GEMM  out[c][m][n] = sum_k A[c][m][k] * B[c][n][k]
// with A/B given as split (hi,lo) f16, K innermost. 256 thr = 8 waves,
// block tile 128x128, wave tile 64x32, K-step 32.
//
// LDS tiles are stored pre-swizzled in exact WMMA per-lane fragment order:
//   A frag (16x32 f16): lane = (m&15) + 16*(kchunk&1), half-sel = kchunk>>1
//   B frag (32x16 f16): lane = (n&15) + 16*(kchunk&1), half-sel = kchunk>>1
// so staging is a per-lane 16B global chunk -> 16B LDS slot, which maps 1:1
// onto GLOBAL_LOAD_ASYNC_TO_LDS_B128. Two LDS buffers, double-buffered on
// ASYNCcnt: issue next tile's copies, s_wait_asynccnt 8 (in-order) for the
// current tile, barrier, 24 WMMAs.
// ---------------------------------------------------------------------------
__global__ __launch_bounds__(256) void gemm_split_f16_wmma_kernel(
    const _Float16* __restrict__ ws, float* __restrict__ out)
{
    // Per buffer (halfs): [0:4096)=Ahi [4096:8192)=Alo [8192:12288)=Bhi [12288:16384)=Blo
    __shared__ _Float16 smem[2 * 16384];   // 2 x 32 KB double buffer

    const int tid  = threadIdx.x;
    const int bx   = blockIdx.x;
    const int c    = blockIdx.y;
    const int m0   = (bx >> 4) * 128;
    const int n0   = (bx & 15) * 128;
    const int wid  = tid >> 5;
    const int lane = tid & 31;
    const int wm   = wid >> 2;   // 0..1 -> 64 rows of M
    const int wn   = wid & 3;    // 0..3 -> 32 cols of N

    const _Float16* Ahi = ws;                  // + OFS -> Alo
    const _Float16* Bhi = ws + 2u * OFS;       // + OFS -> Blo

    // ---- per-thread staging addresses (chunk h=0; h=1 is +64*RANK / +4096B) ----
    const int loc0 = tid >> 2;        // local m (A) / n (B), 0..63
    const int k4   = tid & 3;         // 8-half K chunk id
    const int st0  = loc0 >> 4;       // 16-row subtile
    const int dA0  = ((st0 * 32 + ((loc0 & 15) + ((k4 & 1) << 4))) * 16) + ((k4 >> 1) << 3);
    const int dB0  = ((st0 * 32 + ((loc0 & 15) + ((k4 & 1) << 4))) * 16) + ((k4 >> 1) << 3);
    const _Float16* gA = Ahi + ((size_t)(c * RES + m0 + loc0)) * RANK + k4 * 8;
    const _Float16* gB = Bhi + ((size_t)(c * RES + n0 + loc0)) * RANK + k4 * 8;

    // Low 32 bits of a generic LDS pointer == DS byte address.
    const uint32_t smem0 = (uint32_t)(uintptr_t)(&smem[0]);
    const uint32_t lA0   = smem0 + (uint32_t)dA0 * 2u;            // Ahi slot, buffer 0
    const uint32_t lB0   = smem0 + 16384u + (uint32_t)dB0 * 2u;   // Bhi slot, buffer 0

    auto stage = [&](int buf, int kc) {
        const uint32_t lA = lA0 + (uint32_t)buf * 32768u;
        const uint32_t lB = lB0 + (uint32_t)buf * 32768u;
        const _Float16* pA = gA + kc;
        const _Float16* pB = gB + kc;
        ASYNC_CP16(lA,           pA);                    // Ahi, rows 0..63
        ASYNC_CP16(lA + 4096u,   pA + 64u * RANK);       // Ahi, rows 64..127
        ASYNC_CP16(lA + 8192u,   pA + OFS);              // Alo, rows 0..63
        ASYNC_CP16(lA + 12288u,  pA + OFS + 64u * RANK); // Alo, rows 64..127
        ASYNC_CP16(lB,           pB);                    // Bhi
        ASYNC_CP16(lB + 4096u,   pB + 64u * RANK);
        ASYNC_CP16(lB + 8192u,   pB + OFS);              // Blo
        ASYNC_CP16(lB + 12288u,  pB + OFS + 64u * RANK);
    };

    v8f acc[4][2];
    #pragma unroll
    for (int a = 0; a < 4; ++a)
        #pragma unroll
        for (int b = 0; b < 2; ++b)
            acc[a][b] = (v8f){0.f, 0.f, 0.f, 0.f, 0.f, 0.f, 0.f, 0.f};

    stage(0, 0);   // prologue: 8 async copies into buffer 0

    for (int kc = 0; kc < RANK; kc += 32) {
        const int buf = (kc >> 5) & 1;
        if (kc + 32 < RANK) {
            stage(buf ^ 1, kc + 32);   // prefetch next tile into other buffer
            WAIT_ASYNC(8);             // in-order: current tile's 8 copies landed
        } else {
            WAIT_ASYNC(0);             // last tile: drain everything
        }
        __syncthreads();               // all waves' copies visible in LDS

        const int base = buf * 16384;

        v16h bh[2], bl[2];
        #pragma unroll
        for (int b = 0; b < 2; ++b) {
            int off = base + ((wn * 2 + b) * 32 + lane) * 16;
            bh[b] = *(const v16h*)&smem[8192 + off];
            bl[b] = *(const v16h*)&smem[12288 + off];
        }

        #pragma unroll
        for (int a = 0; a < 4; ++a) {
            int off = base + ((wm * 4 + a) * 32 + lane) * 16;
            v16h ah = *(const v16h*)&smem[off];
            v16h al = *(const v16h*)&smem[4096 + off];
            #pragma unroll
            for (int b = 0; b < 2; ++b) {
                // (hi+lo)x(hi+lo) minus negligible lo*lo term -> ~fp32 product
                acc[a][b] = __builtin_amdgcn_wmma_f32_16x16x32_f16(
                    false, ah, false, bh[b], (short)0, acc[a][b], false, false);
                acc[a][b] = __builtin_amdgcn_wmma_f32_16x16x32_f16(
                    false, al, false, bh[b], (short)0, acc[a][b], false, false);
                acc[a][b] = __builtin_amdgcn_wmma_f32_16x16x32_f16(
                    false, ah, false, bl[b], (short)0, acc[a][b], false, false);
            }
        }
        __syncthreads();               // reads done before this buffer is rewritten
    }

    // Epilogue: C/D layout -> lane(0-15): M=vgpr, lane(16-31): M=vgpr+8; N=lane&15
    const int nl = lane & 15;
    const int mh = (lane >> 4) * 8;
    #pragma unroll
    for (int a = 0; a < 4; ++a) {
        int mbase = m0 + wm * 64 + a * 16 + mh;
        #pragma unroll
        for (int b = 0; b < 2; ++b) {
            int n = n0 + wn * 32 + b * 16 + nl;
            size_t obase = ((size_t)(c * RES + mbase)) * RES + n;
            #pragma unroll
            for (int i = 0; i < 8; ++i)
                out[obase + (size_t)i * RES] = acc[a][b][i];
        }
    }
}

// ---------------------------------------------------------------------------
extern "C" void kernel_launch(void* const* d_in, const int* in_sizes, int n_in,
                              void* d_out, int out_size, void* d_ws, size_t ws_size,
                              hipStream_t stream)
{
    const float* rank1 = (const float*)d_in[0];   // (3,512,2048) fp32
    const float* rank2 = (const float*)d_in[1];   // (3,512,2048) fp32
    const float* kern  = (const float*)d_in[2];   // (51,) fp32
    float*       outp  = (float*)d_out;           // (3,2048,2048) fp32
    _Float16*    W     = (_Float16*)d_ws;         // 4 arrays x OFS halfs = 24 MB

    // Blur + split + transpose both inputs: grid (posChunks, c*rChunks, src)
    blur_split_transpose_kernel<<<dim3(32, 24, 2), 256, 0, stream>>>(
        rank1, rank2, kern, W);

    // 3 channels x (16x16) 128x128 output tiles
    gemm_split_f16_wmma_kernel<<<dim3(256, 3), 256, 0, stream>>>(W, outp);
}
